// GCN_81166291959882
// MI455X (gfx1250) — matrix operand order repaired and checked by reading, not compiled, not run on hardware
//
#include <hip/hip_runtime.h>

#define BN 8
#define NN 2048
#define NH 128
#define TSTEPS 10
#define RR 0.1f
#define NT (NN / 32)

typedef _Float16 h16;
typedef __attribute__((ext_vector_type(16))) _Float16 v16h;
typedef __attribute__((ext_vector_type(8)))  float    v8f;

union F16x8 { uint4 u; h16 h[8]; };
union Frag  { v16h v; uint4 u[2]; h16 h[16]; };

// ---------------------------------------------------------------- init
__global__ __launch_bounds__(256)
void init_kernel(const float* __restrict__ points, float* __restrict__ states,
                 float* __restrict__ traj) {
    int idx = blockIdx.x * blockDim.x + threadIdx.x;
    if (idx >= BN * NN) return;
    const float* p = points + (size_t)idx * 3;
    float* s = states + (size_t)idx * 6;
    float x = p[0], y = p[1], z = p[2];
    s[0] = x; s[1] = y; s[2] = z; s[3] = 0.f; s[4] = 0.f; s[5] = 0.f;
    int b = idx / NN, n = idx % NN;
    float* t0 = traj + ((size_t)b * (TSTEPS + 1)) * NN * 3 + (size_t)n * 3;
    t0[0] = x; t0[1] = y; t0[2] = z;
}

__global__ __launch_bounds__(128)
void convw_kernel(const float* __restrict__ W1, const float* __restrict__ W2,
                  h16* __restrict__ Wt1, h16* __restrict__ Wt2) {
    int idx = blockIdx.x * blockDim.x + threadIdx.x;
    if (idx >= NH * NH) return;
    int k = idx / NH, n = idx % NH;
    Wt1[n * NH + k] = (h16)W1[k * NH + n];
    Wt2[n * NH + k] = (h16)W2[k * NH + n];
}

// ------------------------------------------------- adjacency + degrees
__global__ __launch_bounds__(256)
void adj_kernel(const float* __restrict__ states, const float* __restrict__ pad,
                h16* __restrict__ A, float* __restrict__ dinv) {
    int b = blockIdx.x / NN;
    int i = blockIdx.x % NN;
    int t = threadIdx.x;
    const float* srow = states + ((size_t)b * NN + i) * 6;
    float xi = srow[0], yi = srow[1], zi = srow[2];
    bool padi = pad[(size_t)b * NN + i] > 0.f;
    const float* sb = states + (size_t)b * NN * 6;
    const float* pb = pad + (size_t)b * NN;
    int j0 = t * 8;
    float cnt = 0.f;
    F16x8 out;
#pragma unroll
    for (int u = 0; u < 8; ++u) {
        int j = j0 + u;
        float dx = xi - sb[j * 6 + 0];
        float dy = yi - sb[j * 6 + 1];
        float dz = zi - sb[j * 6 + 2];
        float d2 = dx * dx + dy * dy + dz * dz;
        bool e = (j == i) || ((d2 < RR * RR) && padi && (pb[j] > 0.f));
        float v = e ? 1.f : 0.f;
        cnt += v;
        out.h[u] = (h16)v;
    }
    *(uint4*)(A + ((size_t)b * NN + i) * NN + j0) = out.u;

    __shared__ float red[256];
    red[t] = cnt;
    __syncthreads();
    for (int sft = 128; sft > 0; sft >>= 1) {
        if (t < sft) red[t] += red[t + sft];
        __syncthreads();
    }
    if (t == 0) dinv[(size_t)b * NN + i] = rsqrtf(red[0]);
}

// ------------------------------------------- layer 0: h = (X@W0)*dinv, f-major
__global__ __launch_bounds__(128)
void layer0_kernel(const float* __restrict__ states, const float* __restrict__ W0,
                   const float* __restrict__ dinv, h16* __restrict__ Ht) {
    int b = blockIdx.x / NN;
    int n = blockIdx.x % NN;
    int f = threadIdx.x;
    const float* s = states + ((size_t)b * NN + n) * 6;
    float acc = 0.f;
#pragma unroll
    for (int c = 0; c < 6; ++c) acc += s[c] * W0[c * NH + f];
    acc *= dinv[(size_t)b * NN + n];
    Ht[((size_t)b * NH + f) * NN + n] = (h16)acc;
}

// ---------------- Y = relu( A(NxN) @ Ht^T(Nx128) * dinv + bias ), WMMA f16
// B tiles staged to LDS with async-to-LDS double buffering; A frags reg-double-buffered.
__global__ __launch_bounds__(128)
void agemm_kernel(const h16* __restrict__ A, const h16* __restrict__ Ht,
                  const float* __restrict__ dinv, const float* __restrict__ bias,
                  h16* __restrict__ Y) {
    __shared__ __attribute__((aligned(16))) h16 bs[2][32 * NH]; // 2 x 8KB

    const int MB = NN / 64;
    int b    = blockIdx.x / MB;
    int mblk = (blockIdx.x % MB) * 64;
    int wave = threadIdx.x >> 5;
    int lane = threadIdx.x & 31;
    int lm = lane & 15;
    int hi = lane >> 4;
    int M0 = mblk + wave * 16;
    int m  = M0 + lm;
    int f  = threadIdx.x;               // staging row owned by this thread

    const h16* arow  = A  + ((size_t)b * NN + m) * NN;
    const h16* hrow  = Ht + ((size_t)b * NH + f) * NN;  // this thread's Ht row

    v8f acc[8];
#pragma unroll
    for (int ft = 0; ft < 8; ++ft)
#pragma unroll
        for (int r = 0; r < 8; ++r) acc[ft][r] = 0.f;

    // ---- prologue: stage tile 0 into bs[0], load A frag for tile 0
    {
        unsigned lb = (unsigned)(size_t)(&bs[0][f * 32]);
        unsigned long long gb = (unsigned long long)(size_t)(hrow);
#pragma unroll
        for (int c = 0; c < 4; ++c) {
            unsigned l = lb + c * 16u;
            unsigned long long g = gb + c * 16ull;
            asm volatile("global_load_async_to_lds_b128 %0, %1, off"
                         :: "v"(l), "v"(g) : "memory");
        }
    }
    Frag a;
    a.u[0] = *(const uint4*)(arow + hi * 8);
    a.u[1] = *(const uint4*)(arow + 16 + hi * 8);

    for (int kt = 0; kt < NT; ++kt) {
        int buf = kt & 1;
        Frag an;
        if (kt + 1 < NT) {
            int kn = (kt + 1) * 32;
            unsigned lb = (unsigned)(size_t)(&bs[buf ^ 1][f * 32]);
            unsigned long long gb = (unsigned long long)(size_t)(hrow + kn);
#pragma unroll
            for (int c = 0; c < 4; ++c) {
                unsigned l = lb + c * 16u;
                unsigned long long g = gb + c * 16ull;
                asm volatile("global_load_async_to_lds_b128 %0, %1, off"
                             :: "v"(l), "v"(g) : "memory");
            }
            an.u[0] = *(const uint4*)(arow + kn + hi * 8);
            an.u[1] = *(const uint4*)(arow + kn + 16 + hi * 8);
            asm volatile("s_wait_asynccnt 0x4" ::: "memory"); // prev tile staged
        } else {
            asm volatile("s_wait_asynccnt 0x0" ::: "memory"); // last tile staged
        }
        __syncthreads(); // tile kt visible to all waves

        Frag bf[8];
#pragma unroll
        for (int ft = 0; ft < 8; ++ft) {
            const uint4* p = (const uint4*)(&bs[buf][(ft * 16 + lm) * 32 + hi * 8]);
            bf[ft].u[0] = p[0];
            bf[ft].u[1] = p[2];
        }
#pragma unroll
        for (int ft = 0; ft < 8; ++ft)
            acc[ft] = __builtin_amdgcn_wmma_f32_16x16x32_f16(
                false, a.v, false, bf[ft].v, (short)0, acc[ft], false, false);

        __syncthreads(); // everyone done reading bs[buf] before it is restaged
        a = an;
    }

    // ---- epilogue: scale/bias/relu, transpose via private LDS quarter, coalesced stores
    float dv[8];
#pragma unroll
    for (int r = 0; r < 8; ++r) dv[r] = dinv[(size_t)b * NN + M0 + hi * 8 + r];

    h16* myT = &bs[0][0] + (size_t)wave * (16 * NH); // 4KB private per wave
#pragma unroll
    for (int ft = 0; ft < 8; ++ft) {
        int fo = ft * 16 + lm;
        float bia = bias[fo];
#pragma unroll
        for (int r = 0; r < 8; ++r) {
            float v = acc[ft][r] * dv[r] + bia;
            v = v > 0.f ? v : 0.f;
            myT[(r + hi * 8) * NH + fo] = (h16)v;
        }
    }
    // lane packs 64 contiguous halves (half a row) -> 8 b128 stores
    {
        const uint4* src = (const uint4*)(myT + lane * 64);
        int mr  = M0 + (lane >> 1);
        int col = (lane & 1) * 64;
        uint4* dst = (uint4*)(Y + ((size_t)b * NN + mr) * NH + col);
#pragma unroll
        for (int c = 0; c < 8; ++c) dst[c] = src[c];
    }
}

// ------------- h = (Y(Nx128) @ W(128x128)) * dinv, output feature-major
__global__ __launch_bounds__(128)
void xw_kernel(const h16* __restrict__ X, const h16* __restrict__ Wt,
               const float* __restrict__ dinv, h16* __restrict__ Ht) {
    const int MB = NN / 64;
    int b    = blockIdx.x / MB;
    int mblk = (blockIdx.x % MB) * 64;
    int wave = threadIdx.x >> 5;
    int lane = threadIdx.x & 31;
    int lm = lane & 15;
    int hi = lane >> 4;
    int M0 = mblk + wave * 16;
    int m  = M0 + lm;

    const h16* xrow = X + ((size_t)b * NN + m) * NH;

    v8f acc[8];
#pragma unroll
    for (int ft = 0; ft < 8; ++ft)
#pragma unroll
        for (int r = 0; r < 8; ++r) acc[ft][r] = 0.f;

#pragma unroll
    for (int kb = 0; kb < NH; kb += 32) {
        Frag a;
        a.u[0] = *(const uint4*)(xrow + kb + hi * 8);
        a.u[1] = *(const uint4*)(xrow + kb + 16 + hi * 8);
        Frag bf[8];
#pragma unroll
        for (int ft = 0; ft < 8; ++ft) {
            const h16* wrow = Wt + (size_t)(ft * 16 + lm) * NH + kb + hi * 8;
            bf[ft].u[0] = *(const uint4*)(wrow);
            bf[ft].u[1] = *(const uint4*)(wrow + 16);
        }
#pragma unroll
        for (int ft = 0; ft < 8; ++ft)
            acc[ft] = __builtin_amdgcn_wmma_f32_16x16x32_f16(
                false, a.v, false, bf[ft].v, (short)0, acc[ft], false, false);
    }

    float dv[8];
#pragma unroll
    for (int r = 0; r < 8; ++r) dv[r] = dinv[(size_t)b * NN + M0 + hi * 8 + r];
#pragma unroll
    for (int ft = 0; ft < 8; ++ft) {
        int fo = ft * 16 + lm;
        F16x8 o;
#pragma unroll
        for (int r = 0; r < 8; ++r) o.h[r] = (h16)(acc[ft][r] * dv[r]);
        *(uint4*)(Ht + ((size_t)b * NH + fo) * NN + M0 + hi * 8) = o.u;
    }
}

// ----------------- res = Y@Wfc + bfc ; state += res*pad ; write trajectory
__global__ __launch_bounds__(256)
void fc_kernel(const h16* __restrict__ Y, const float* __restrict__ Wfc,
               const float* __restrict__ bfc, const float* __restrict__ pad,
               float* __restrict__ states, float* __restrict__ traj, int tstep) {
    int idx = blockIdx.x * blockDim.x + threadIdx.x;
    if (idx >= BN * NN) return;
    int b = idx / NN, n = idx % NN;
    const h16* y = Y + (size_t)idx * NH;
    float res[6];
#pragma unroll
    for (int c = 0; c < 6; ++c) res[c] = bfc[c];
    for (int t = 0; t < NH; ++t) {
        float yv = (float)y[t];
#pragma unroll
        for (int c = 0; c < 6; ++c) res[c] += yv * Wfc[t * 6 + c];
    }
    float pv = pad[idx];
    float* s = states + (size_t)idx * 6;
    float ns[6];
#pragma unroll
    for (int c = 0; c < 6; ++c) { ns[c] = s[c] + res[c] * pv; s[c] = ns[c]; }
    float* tp = traj + ((size_t)b * (TSTEPS + 1) + tstep + 1) * NN * 3 + (size_t)n * 3;
    tp[0] = ns[0]; tp[1] = ns[1]; tp[2] = ns[2];
}

extern "C" void kernel_launch(void* const* d_in, const int* in_sizes, int n_in,
                              void* d_out, int out_size, void* d_ws, size_t ws_size,
                              hipStream_t stream) {
    const float* points = (const float*)d_in[0];
    const float* pad    = (const float*)d_in[5];
    const float* W0  = (const float*)d_in[6];
    const float* b0  = (const float*)d_in[7];
    const float* W1  = (const float*)d_in[8];
    const float* b1  = (const float*)d_in[9];
    const float* W2  = (const float*)d_in[10];
    const float* b2  = (const float*)d_in[11];
    const float* Wfc = (const float*)d_in[12];
    const float* bfc = (const float*)d_in[13];
    float* traj = (float*)d_out;

    char* ws = (char*)d_ws;
    size_t off = 0;
    auto alloc = [&](size_t bytes) -> void* {
        void* p = ws + off;
        off += (bytes + 255) & ~(size_t)255;
        return p;
    };
    float* states = (float*)alloc((size_t)BN * NN * 6 * sizeof(float));
    h16*   A      = (h16*)  alloc((size_t)BN * NN * NN * sizeof(h16));
    float* dinv   = (float*)alloc((size_t)BN * NN * sizeof(float));
    h16*   Ht     = (h16*)  alloc((size_t)BN * NH * NN * sizeof(h16));
    h16*   Y      = (h16*)  alloc((size_t)BN * NN * NH * sizeof(h16));
    h16*   Wt1    = (h16*)  alloc((size_t)NH * NH * sizeof(h16));
    h16*   Wt2    = (h16*)  alloc((size_t)NH * NH * sizeof(h16));

    init_kernel<<<(BN * NN + 255) / 256, 256, 0, stream>>>(points, states, traj);
    convw_kernel<<<(NH * NH + 127) / 128, 128, 0, stream>>>(W1, W2, Wt1, Wt2);

    const int gGemm = BN * (NN / 64);
    for (int t = 0; t < TSTEPS; ++t) {
        adj_kernel   <<<BN * NN, 256, 0, stream>>>(states, pad, A, dinv);
        layer0_kernel<<<BN * NN, 128, 0, stream>>>(states, W0, dinv, Ht);
        agemm_kernel <<<gGemm, 128, 0, stream>>>(A, Ht, dinv, b0, Y);
        xw_kernel    <<<gGemm, 128, 0, stream>>>(Y, Wt1, dinv, Ht);
        agemm_kernel <<<gGemm, 128, 0, stream>>>(A, Ht, dinv, b1, Y);
        xw_kernel    <<<gGemm, 128, 0, stream>>>(Y, Wt2, dinv, Ht);
        agemm_kernel <<<gGemm, 128, 0, stream>>>(A, Ht, dinv, b2, Y);
        fc_kernel    <<<(BN * NN + 255) / 256, 256, 0, stream>>>(Y, Wfc, bfc, pad,
                                                                 states, traj, t);
    }
}
